// GATLayer_42262478193225
// MI455X (gfx1250) — matrix operand order repaired and checked by reading, not compiled, not run on hardware
//
#include <hip/hip_runtime.h>
#include <cmath>

// ---------------------------------------------------------------------------
// GAT x3 + BN + ReLU + residual for MI455X (gfx1250, wave32, WMMA).
// Dense projections: v_wmma_f32_16x16x32_f16, K=128 fully unrolled, B
// fragments register-resident across 5 row tiles per wave.
// Sparse softmax-aggregation: native f32 global atomics (3-pass), float4
// vectorized scatter in pass 3.
// ---------------------------------------------------------------------------

typedef __attribute__((ext_vector_type(16))) _Float16 v16h;
typedef __attribute__((ext_vector_type(8)))  float    v8f;

#define TPB 256
#define EPS 1e-5f

// -------------------------- fill ------------------------------------------
__global__ void fill_f32(float* __restrict__ p, float v, long long n) {
  long long i = (long long)blockIdx.x * blockDim.x + threadIdx.x;
  if (i < n) p[i] = v;
}

// -------------------------- WMMA GEMM --------------------------------------
// C[M,Nout] = A[M,K=128] * W[Nout,K=128]^T      (h = x @ W.T)
// grid = (Mtiles/MT, Ntiles/4), block = (32,4): one wave32 per (MT x 1) strip
// of 16x16 output tiles.  B fragments (4 k-steps) held in registers and
// reused across MT row tiles.
// A and B fragments share contiguous k order (k = 16*lane_half + element),
// C/D per the 32-bit 16x16 table.
template <int MT>
__global__ void gemm_wmma_f16(const float* __restrict__ A,
                              const float* __restrict__ W,
                              float* __restrict__ C, int Nout) {
  constexpr int K = 128;
  const int lane = threadIdx.x;                            // 0..31
  const int nt   = blockIdx.y * blockDim.y + threadIdx.y;  // col tile
  const int mt0  = blockIdx.x * MT;                        // first row tile
  const int l15  = lane & 15;
  const int kg   = lane >> 4;                              // lane half

  // ---- load all B fragments once (W row = output column) ----
  const float2* W2 = (const float2*)(W + (size_t)(nt * 16 + l15) * K);
  v16h b[4];
#pragma unroll
  for (int s = 0; s < 4; ++s) {
#pragma unroll
    for (int i = 0; i < 8; ++i) {
      // B: lanes 0-15 -> K 0..15 ; lanes 16-31 -> K 16..31 (2 per VGPR)
      int kb = 32 * s + kg * 16 + 2 * i;
      float2 fb = W2[kb >> 1];
      b[s][2 * i]     = (_Float16)fb.x;
      b[s][2 * i + 1] = (_Float16)fb.y;
    }
  }

#pragma unroll
  for (int t = 0; t < MT; ++t) {
    const float2* A2 = (const float2*)(A + (size_t)((mt0 + t) * 16 + l15) * K);
    v8f acc = {};
#pragma unroll
    for (int s = 0; s < 4; ++s) {
      v16h a;
#pragma unroll
      for (int i = 0; i < 8; ++i) {
        // A: lanes 0-15 -> K 0..15 ; lanes 16-31 -> K 16..31 (2 per VGPR)
        int ka = 32 * s + kg * 16 + 2 * i;
        float2 fa = A2[ka >> 1];
        a[2 * i]     = (_Float16)fa.x;
        a[2 * i + 1] = (_Float16)fa.y;
      }
      acc = __builtin_amdgcn_wmma_f32_16x16x32_f16(false, a, false, b[s],
                                                   (short)0, acc, false, false);
    }
    // D: VGPR r -> row r (lanes 0-15) / r+8 (lanes 16-31); col = lane&15
    float* Cp = C + (size_t)((mt0 + t) * 16 + kg * 8) * Nout + nt * 16 + l15;
#pragma unroll
    for (int r = 0; r < 8; ++r) Cp[(size_t)r * Nout] = acc[r];
  }
}

// -------------------- attention coefficients -------------------------------
// es[n,h] = sum_c h[n,h,c]*a_s[h,c] ; ed likewise.  One thread per (node,head).
__global__ void attn_coeffs(const float* __restrict__ h,
                            const float* __restrict__ as_,
                            const float* __restrict__ ad_,
                            float* __restrict__ es, float* __restrict__ ed,
                            int n, int C) {
  int i = blockIdx.x * blockDim.x + threadIdx.x;
  if (i >= n * 2) return;
  int node = i >> 1, head = i & 1;
  const float* hp = h + (size_t)node * (2 * C) + head * C;
  const float* ap = as_ + head * C;
  const float* bp = ad_ + head * C;
  float s = 0.f, d = 0.f;
  for (int c = 0; c < C; ++c) {
    float v = hp[c];
    s = fmaf(v, ap[c], s);
    d = fmaf(v, bp[c], d);
  }
  es[i] = s;
  ed[i] = d;
}

// ---------------- pass 1: leaky-relu logits + segment max ------------------
__global__ void edge_pass1(const int* __restrict__ src, const int* __restrict__ dst,
                           const float* __restrict__ es, const float* __restrict__ ed,
                           float* __restrict__ lbuf, int* __restrict__ mx, int EH) {
  int i = blockIdx.x * blockDim.x + threadIdx.x;
  if (i >= EH) return;
  int e = i >> 1, head = i & 1;
  int s = src[e], d = dst[e];
  float v = es[s * 2 + head] + ed[d * 2 + head];
  v = (v > 0.f) ? v : 0.2f * v;                 // leaky_relu slope 0.2
  lbuf[i] = v;
  int* addr = mx + d * 2 + head;                // float max via int atomics
  if (v >= 0.f) atomicMax(addr, __float_as_int(v));
  else          atomicMin((unsigned int*)addr, __float_as_uint(v));
}

// ---------------- pass 2: exp + segment sum --------------------------------
__global__ void edge_pass2(const int* __restrict__ dst,
                           float* __restrict__ lbuf, const int* __restrict__ mx,
                           float* __restrict__ den, int EH) {
  int i = blockIdx.x * blockDim.x + threadIdx.x;
  if (i >= EH) return;
  int e = i >> 1, head = i & 1;
  int d = dst[e];
  float m = __int_as_float(mx[d * 2 + head]);
  float ex = __expf(lbuf[i] - m);
  lbuf[i] = ex;
  atomicAdd(&den[d * 2 + head], ex);
}

// ---------------- pass 3: alpha-weighted scatter-add -----------------------
// One thread per (edge, 4 channels): one float4 gather of h[src], one alpha,
// four global f32 atomic adds.  CH = 2*C is 128 or 256 (power of two).
__global__ void edge_pass3(const int* __restrict__ src, const int* __restrict__ dst,
                           const float* __restrict__ h, const float* __restrict__ exb,
                           const float* __restrict__ den, float* __restrict__ agg,
                           long long total4, int chShift) {
  long long i = (long long)blockIdx.x * blockDim.x + threadIdx.x;
  if (i >= total4) return;
  const int perEdgeShift = chShift - 2;               // CH/4 groups per edge
  int e = (int)(i >> perEdgeShift);
  int r = ((int)(i & ((1 << perEdgeShift) - 1))) << 2;  // base channel
  int C = 1 << (chShift - 1);
  int head = (r >= C) ? 1 : 0;
  int s = src[e], d = dst[e];
  float alpha = exb[e * 2 + head] / (den[d * 2 + head] + 1e-16f);
  int CH = 1 << chShift;
  const float4 hv = *(const float4*)(h + (size_t)s * CH + r);
  float* ap = agg + (size_t)d * CH + r;
  atomicAdd(ap + 0, hv.x * alpha);
  atomicAdd(ap + 1, hv.y * alpha);
  atomicAdd(ap + 2, hv.z * alpha);
  atomicAdd(ap + 3, hv.w * alpha);
}

// ---------------- fused bias + BN(eval) + ReLU -----------------------------
__global__ void bias_bn_relu(const float* __restrict__ gin, const float* __restrict__ b,
                             const float* __restrict__ g, const float* __restrict__ be,
                             const float* __restrict__ m, const float* __restrict__ v,
                             float* __restrict__ out, long long total, int cMask) {
  long long i = (long long)blockIdx.x * blockDim.x + threadIdx.x;
  if (i >= total) return;
  int c = (int)(i & cMask);
  float x = gin[i] + b[c];
  x = (x - m[c]) * rsqrtf(v[c] + EPS) * g[c] + be[c];
  out[i] = fmaxf(x, 0.f);
}

// ------- final: head-mean + bias + BN + ReLU + residual (+br) --------------
__global__ void final_mean_bn_res(const float* __restrict__ agg, const float* __restrict__ b3,
                                  const float* __restrict__ g, const float* __restrict__ be,
                                  const float* __restrict__ m, const float* __restrict__ v,
                                  const float* __restrict__ res, const float* __restrict__ br,
                                  float* __restrict__ out, long long total) {
  long long i = (long long)blockIdx.x * blockDim.x + threadIdx.x;
  if (i >= total) return;
  long long node = i >> 7;
  int c = (int)(i & 127);
  float x = 0.5f * (agg[node * 256 + c] + agg[node * 256 + 128 + c]) + b3[c];
  x = (x - m[c]) * rsqrtf(v[c] + EPS) * g[c] + be[c];
  out[i] = fmaxf(x, 0.f) + res[i] + br[c];
}

// ---------------------------------------------------------------------------
static inline unsigned int blocks_for(long long n) {
  return (unsigned int)((n + TPB - 1) / TPB);
}

extern "C" void kernel_launch(void* const* d_in, const int* in_sizes, int n_in,
                              void* d_out, int out_size, void* d_ws, size_t ws_size,
                              hipStream_t stream) {
  const float* x   = (const float*)d_in[0];
  const int*   ei  = (const int*)d_in[1];
  const float* W1  = (const float*)d_in[2];
  const float* a1s = (const float*)d_in[3];
  const float* a1d = (const float*)d_in[4];
  const float* b1  = (const float*)d_in[5];
  const float* W2  = (const float*)d_in[6];
  const float* a2s = (const float*)d_in[7];
  const float* a2d = (const float*)d_in[8];
  const float* b2  = (const float*)d_in[9];
  const float* W3  = (const float*)d_in[10];
  const float* a3s = (const float*)d_in[11];
  const float* a3d = (const float*)d_in[12];
  const float* b3  = (const float*)d_in[13];
  const float* g1  = (const float*)d_in[14];
  const float* be1 = (const float*)d_in[15];
  const float* m1  = (const float*)d_in[16];
  const float* v1  = (const float*)d_in[17];
  const float* g2  = (const float*)d_in[18];
  const float* be2 = (const float*)d_in[19];
  const float* m2  = (const float*)d_in[20];
  const float* v2  = (const float*)d_in[21];
  const float* g3  = (const float*)d_in[22];
  const float* be3 = (const float*)d_in[23];
  const float* m3  = (const float*)d_in[24];
  const float* v3  = (const float*)d_in[25];
  const float* Wr  = (const float*)d_in[26];
  const float* br  = (const float*)d_in[27];

  const int N = in_sizes[0] / 128;   // 50000  (Mtiles = 3125 = 5 * 625)
  const int E = in_sizes[1] / 2;     // 800000
  const int* src = ei;
  const int* dst = ei + E;

  // -------- workspace carve-up (floats) --------
  float* ws = (float*)d_ws;
  size_t o = 0;
  float* G   = ws + o; o += (size_t)N * 256;   // GEMM output h (per-head feats)
  float* AGG = ws + o; o += (size_t)N * 256;   // segment-sum aggregation
  float* F0  = ws + o; o += (size_t)N * 128;   // inter-layer features
  float* RES = ws + o; o += (size_t)N * 128;   // residual branch
  float* ES  = ws + o; o += (size_t)N * 2;
  float* ED  = ws + o; o += (size_t)N * 2;
  float* MX  = ws + o; o += (size_t)N * 2;     // segment max (float bits)
  float* DEN = ws + o; o += (size_t)N * 2;
  float* EXB = ws + o; o += (size_t)E * 2;     // per-edge logits -> exp

  const int  EH   = E * 2;
  const long long N128 = (long long)N * 128;
  const long long N256 = (long long)N * 256;
  constexpr int MT = 5;                          // row tiles per wave
  const dim3 gblk(32, 4);                        // 4 wave32 per block
  const dim3 ggrid128(N / 16 / MT, 128 / 16 / 4);  // Nout = 128
  const dim3 ggrid256(N / 16 / MT, 256 / 16 / 4);  // Nout = 256
  const float NEG_INF = -__builtin_huge_valf();

  // ---- residual branch: RES = x @ Wr.T (br folded into final kernel) ----
  gemm_wmma_f16<MT><<<ggrid128, gblk, 0, stream>>>(x, Wr, RES, 128);

  // ================= layer 1 (in=x, C=64, concat) =================
  gemm_wmma_f16<MT><<<ggrid128, gblk, 0, stream>>>(x, W1, G, 128);
  attn_coeffs<<<blocks_for(N * 2), TPB, 0, stream>>>(G, a1s, a1d, ES, ED, N, 64);
  fill_f32<<<blocks_for(N * 2), TPB, 0, stream>>>(MX, NEG_INF, N * 2);
  fill_f32<<<blocks_for(N * 2), TPB, 0, stream>>>(DEN, 0.f, N * 2);
  fill_f32<<<blocks_for(N128), TPB, 0, stream>>>(AGG, 0.f, N128);
  edge_pass1<<<blocks_for(EH), TPB, 0, stream>>>(src, dst, ES, ED, EXB, (int*)MX, EH);
  edge_pass2<<<blocks_for(EH), TPB, 0, stream>>>(dst, EXB, (const int*)MX, DEN, EH);
  edge_pass3<<<blocks_for((long long)E * 32), TPB, 0, stream>>>(
      src, dst, G, EXB, DEN, AGG, (long long)E * 32, 7);
  bias_bn_relu<<<blocks_for(N128), TPB, 0, stream>>>(AGG, b1, g1, be1, m1, v1, F0, N128, 127);

  // ================= layer 2 (in=F0, C=64, concat) =================
  gemm_wmma_f16<MT><<<ggrid128, gblk, 0, stream>>>(F0, W2, G, 128);
  attn_coeffs<<<blocks_for(N * 2), TPB, 0, stream>>>(G, a2s, a2d, ES, ED, N, 64);
  fill_f32<<<blocks_for(N * 2), TPB, 0, stream>>>(MX, NEG_INF, N * 2);
  fill_f32<<<blocks_for(N * 2), TPB, 0, stream>>>(DEN, 0.f, N * 2);
  fill_f32<<<blocks_for(N128), TPB, 0, stream>>>(AGG, 0.f, N128);
  edge_pass1<<<blocks_for(EH), TPB, 0, stream>>>(src, dst, ES, ED, EXB, (int*)MX, EH);
  edge_pass2<<<blocks_for(EH), TPB, 0, stream>>>(dst, EXB, (const int*)MX, DEN, EH);
  edge_pass3<<<blocks_for((long long)E * 32), TPB, 0, stream>>>(
      src, dst, G, EXB, DEN, AGG, (long long)E * 32, 7);
  bias_bn_relu<<<blocks_for(N128), TPB, 0, stream>>>(AGG, b2, g2, be2, m2, v2, F0, N128, 127);

  // ============ layer 3 (in=F0, C=128, mean over heads) ============
  gemm_wmma_f16<MT><<<ggrid256, gblk, 0, stream>>>(F0, W3, G, 256);
  attn_coeffs<<<blocks_for(N * 2), TPB, 0, stream>>>(G, a3s, a3d, ES, ED, N, 128);
  fill_f32<<<blocks_for(N * 2), TPB, 0, stream>>>(MX, NEG_INF, N * 2);
  fill_f32<<<blocks_for(N * 2), TPB, 0, stream>>>(DEN, 0.f, N * 2);
  fill_f32<<<blocks_for(N256), TPB, 0, stream>>>(AGG, 0.f, N256);
  edge_pass1<<<blocks_for(EH), TPB, 0, stream>>>(src, dst, ES, ED, EXB, (int*)MX, EH);
  edge_pass2<<<blocks_for(EH), TPB, 0, stream>>>(dst, EXB, (const int*)MX, DEN, EH);
  edge_pass3<<<blocks_for((long long)E * 64), TPB, 0, stream>>>(
      src, dst, G, EXB, DEN, AGG, (long long)E * 64, 8);
  final_mean_bn_res<<<blocks_for(N128), TPB, 0, stream>>>(
      AGG, b3, g3, be3, m3, v3, RES, br, (float*)d_out, N128);
}
